// NeuralCDEModel_10883447128346
// MI455X (gfx1250) — compile-verified, matrix operand order checked by reading, not run on hardware
//
#include <hip/hip_runtime.h>
#include <hip/hip_bf16.h>
#include <math.h>

// ---------------------------------------------------------------------------
// Problem dims
// ---------------------------------------------------------------------------
#define BATCH      8192
#define IN_DIM     1024
#define PATH_DIM   256
#define HID_DIM    1024
#define OUT_DIM    1024
#define FUSE_DIM   (HID_DIM + PATH_DIM)   // 1280
#define NSTEPS     50

// GEMM tiling
#define BM 128
#define BN 128
#define BK 32
#define LDSS 40          // 32 + 8 pad (row stride = 80B = 5*16B, keeps b128 aligned)

typedef __bf16 bf16x16 __attribute__((ext_vector_type(16)));
typedef float  f32x8   __attribute__((ext_vector_type(8)));

union FragAB { bf16x16 v; uint4 q[2]; };

__device__ __forceinline__ unsigned short f2bf(float f) {
  union { float f; unsigned int u; } cv; cv.f = f;
  unsigned int u = cv.u;
  unsigned int r = u + 0x7FFFu + ((u >> 16) & 1u);   // round-nearest-even
  return (unsigned short)(r >> 16);
}

// CDNA5 async global->LDS copy (ASYNCcnt-tracked, no VGPR round trip).
// ldsoff = byte offset within workgroup LDS allocation (low 32 bits of
// the generic pointer per the LDS aperture truncation rule).
__device__ __forceinline__ void async_copy_b128(unsigned ldsoff,
                                                const void* gptr) {
  asm volatile("global_load_async_to_lds_b128 %0, %1, off"
               :: "v"(ldsoff), "v"(gptr)
               : "memory");
}
__device__ __forceinline__ void wait_async0() {
  asm volatile("s_wait_asynccnt 0x0" ::: "memory");
}

// ---------------------------------------------------------------------------
// Prep kernels
// ---------------------------------------------------------------------------
__global__ void k_f32_to_bf16(const float* __restrict__ in,
                              unsigned short* __restrict__ out, int n) {
  int i = blockIdx.x * 256 + threadIdx.x;
  if (i < n) out[i] = f2bf(in[i]);
}

// W is [K][N] row-major fp32 -> Wt is [N][K] row-major bf16
__global__ void k_transpose_bf16(const float* __restrict__ W,
                                 unsigned short* __restrict__ Wt,
                                 int K, int N) {
  int i = blockIdx.x * 256 + threadIdx.x;
  if (i < K * N) {
    int k = i / N, n = i - k * N;
    Wt[(size_t)n * K + k] = f2bf(W[i]);
  }
}

// inp[:, 1024:1280] = bf16(t * g)
__global__ void k_scale_g(const float* __restrict__ g,
                          unsigned short* __restrict__ inp, float t) {
  int i = blockIdx.x * 256 + threadIdx.x;     // over BATCH*PATH_DIM
  int r = i >> 8, c = i & 255;
  inp[(size_t)r * FUSE_DIM + HID_DIM + c] = f2bf(t * g[i]);
}

// ---------------------------------------------------------------------------
// bf16 WMMA GEMM:  C[M,N] = A[M,K](bf16, lda) * Bt[N,K](bf16, ldb) + bias
// Double-buffered LDS, async global->LDS staging, wave32 WMMA compute.
// epilogue modes:
//   0: outf = val                         (g, final readout)
//   1: z = val; outbf = bf16(val)         (z0 init, mirrored into inp)
//   2: outbf = bf16(tanh(val))            (hidden h)
//   3: z += dt*val; outbf = bf16(z)       (step update, mirrored into inp)
// ---------------------------------------------------------------------------
__global__ __launch_bounds__(256)
void k_gemm_bf16_wmma(const unsigned short* __restrict__ A, int lda,
                      const unsigned short* __restrict__ Bt, int ldb,
                      const float* __restrict__ bias,
                      int M, int N, int K,
                      int mode, float dt,
                      float* __restrict__ zbuf,
                      float* __restrict__ outf,
                      unsigned short* __restrict__ outbf, int ldo) {
  __shared__ unsigned short lds_a[2][BM * LDSS];
  __shared__ unsigned short lds_b[2][BN * LDSS];

  const int tid   = threadIdx.x;
  const int lane  = tid & 31;
  const int wave  = tid >> 5;
  const int waveM = wave & 1;       // 2 waves along M  -> 64 rows each
  const int waveN = wave >> 1;      // 4 waves along N  -> 32 cols each
  const int mlane = lane & 15;
  const int half  = lane >> 4;

  const int blockM = blockIdx.x * BM;
  const int blockN = blockIdx.y * BN;

  // staging: tile = 128 rows x 4 uint4 = 512 uint4; 256 threads x 2 each
  const int r0 = tid >> 2,         c0 = tid & 3;
  const int r1 = (tid + 256) >> 2, c1 = (tid + 256) & 3;

  // per-thread global base pointers (element units)
  const unsigned short* gA0 = A  + (size_t)(blockM + r0) * lda + c0 * 8;
  const unsigned short* gA1 = A  + (size_t)(blockM + r1) * lda + c1 * 8;
  const unsigned short* gB0 = Bt + (size_t)(blockN + r0) * ldb + c0 * 8;
  const unsigned short* gB1 = Bt + (size_t)(blockN + r1) * ldb + c1 * 8;

  // per-thread LDS destination offsets for each buffer
  unsigned la0[2], la1[2], lb0[2], lb1[2];
#pragma unroll
  for (int b = 0; b < 2; ++b) {
    la0[b] = (unsigned)(uintptr_t)&lds_a[b][r0 * LDSS + c0 * 8];
    la1[b] = (unsigned)(uintptr_t)&lds_a[b][r1 * LDSS + c1 * 8];
    lb0[b] = (unsigned)(uintptr_t)&lds_b[b][r0 * LDSS + c0 * 8];
    lb1[b] = (unsigned)(uintptr_t)&lds_b[b][r1 * LDSS + c1 * 8];
  }

  auto issue_tile = [&](int buf, int kt) {
    async_copy_b128(la0[buf], gA0 + kt);
    async_copy_b128(la1[buf], gA1 + kt);
    async_copy_b128(lb0[buf], gB0 + kt);
    async_copy_b128(lb1[buf], gB1 + kt);
  };

  f32x8 acc[4][2] = {};

  // prologue: stage tile 0 into buffer 0
  issue_tile(0, 0);

  const int nk = K / BK;
  for (int it = 0; it < nk; ++it) {
    const int buf = it & 1;

    // drain own async copies for this tile, then make whole-tile visible
    wait_async0();
    __syncthreads();

    // overlap: stage next tile into the other buffer while computing
    if (it + 1 < nk) issue_tile(buf ^ 1, (it + 1) * BK);
    if (it + 2 < nk) {   // warm L2 two tiles ahead -> global_prefetch_b8
      __builtin_prefetch(gA0 + (it + 2) * BK, 0, 1);
      __builtin_prefetch(gB0 + (it + 2) * BK, 0, 1);
    }

    // A fragment (16x32 bf16): lane m = mlane, VGPR0-3: K = half*8+0..7,
    // VGPR4-7: K = 16+half*8+0..7  -> two 16B LDS reads per fragment.
    FragAB af[4], bfr[2];
#pragma unroll
    for (int fm = 0; fm < 4; ++fm) {
      const unsigned short* p =
          &lds_a[buf][(waveM * 64 + fm * 16 + mlane) * LDSS + half * 8];
      af[fm].q[0] = *(const uint4*)(p);
      af[fm].q[1] = *(const uint4*)(p + 16);
    }
#pragma unroll
    for (int fn = 0; fn < 2; ++fn) {
      const unsigned short* p =
          &lds_b[buf][(waveN * 32 + fn * 16 + mlane) * LDSS + half * 8];
      bfr[fn].q[0] = *(const uint4*)(p);
      bfr[fn].q[1] = *(const uint4*)(p + 16);
    }

#pragma unroll
    for (int fm = 0; fm < 4; ++fm)
#pragma unroll
      for (int fn = 0; fn < 2; ++fn)
        acc[fm][fn] = __builtin_amdgcn_wmma_f32_16x16x32_bf16(
            false, af[fm].v, false, bfr[fn].v,
            (short)0, acc[fm][fn], false, false);
  }

  // Epilogue — C/D layout: lane n = mlane, VGPR v -> row = v + 8*half
#pragma unroll
  for (int fm = 0; fm < 4; ++fm) {
#pragma unroll
    for (int fn = 0; fn < 2; ++fn) {
      const int c = blockN + waveN * 32 + fn * 16 + mlane;
      const float bv = bias[c];
#pragma unroll
      for (int v = 0; v < 8; ++v) {
        const int r = blockM + waveM * 64 + fm * 16 + half * 8 + v;
        const float val = acc[fm][fn][v] + bv;
        if (mode == 0) {
          outf[(size_t)r * N + c] = val;
        } else if (mode == 1) {
          zbuf[(size_t)r * N + c] = val;
          outbf[(size_t)r * ldo + c] = f2bf(val);
        } else if (mode == 2) {
          outbf[(size_t)r * ldo + c] = f2bf(tanhf(val));
        } else {
          const size_t zi = (size_t)r * N + c;
          const float zn = zbuf[zi] + dt * val;
          zbuf[zi] = zn;
          outbf[(size_t)r * ldo + c] = f2bf(zn);
        }
      }
    }
  }
}

// ---------------------------------------------------------------------------
// Host launcher
// ---------------------------------------------------------------------------
extern "C" void kernel_launch(void* const* d_in, const int* in_sizes, int n_in,
                              void* d_out, int out_size, void* d_ws, size_t ws_size,
                              hipStream_t stream) {
  (void)in_sizes; (void)n_in; (void)out_size; (void)ws_size;

  const float* x0   = (const float*)d_in[0];
  const float* W_pe = (const float*)d_in[1];
  const float* b_pe = (const float*)d_in[2];
  const float* W_hi = (const float*)d_in[3];
  const float* b_hi = (const float*)d_in[4];
  const float* W1   = (const float*)d_in[5];
  const float* b1   = (const float*)d_in[6];
  const float* W2   = (const float*)d_in[7];
  const float* b2   = (const float*)d_in[8];
  const float* W_ro = (const float*)d_in[9];
  const float* b_ro = (const float*)d_in[10];
  float* out = (float*)d_out;

  // workspace carve-up (256B aligned)
  char* ws = (char*)d_ws;
  size_t off = 0;
  auto carve = [&](size_t bytes) {
    char* p = ws + off;
    off += (bytes + 255) & ~(size_t)255;
    return p;
  };
  unsigned short* x0_bf  = (unsigned short*)carve((size_t)BATCH * IN_DIM * 2);
  unsigned short* wpe_t  = (unsigned short*)carve((size_t)PATH_DIM * IN_DIM * 2);
  unsigned short* whi_t  = (unsigned short*)carve((size_t)HID_DIM * IN_DIM * 2);
  unsigned short* w1_t   = (unsigned short*)carve((size_t)HID_DIM * FUSE_DIM * 2);
  unsigned short* w2_t   = (unsigned short*)carve((size_t)HID_DIM * HID_DIM * 2);
  unsigned short* wro_t  = (unsigned short*)carve((size_t)OUT_DIM * HID_DIM * 2);
  float*          gbuf   = (float*)carve((size_t)BATCH * PATH_DIM * 4);
  float*          zbuf   = (float*)carve((size_t)BATCH * HID_DIM * 4);
  unsigned short* inp_bf = (unsigned short*)carve((size_t)BATCH * FUSE_DIM * 2);
  unsigned short* h_bf   = (unsigned short*)carve((size_t)BATCH * HID_DIM * 2);

  const float dt = 1.0f / (float)NSTEPS;
  const dim3 blk(256);

  // ---- one-time conversions ----
  {
    int n = BATCH * IN_DIM;
    k_f32_to_bf16<<<dim3((n + 255) / 256), blk, 0, stream>>>(x0, x0_bf, n);
  }
  k_transpose_bf16<<<dim3((IN_DIM * PATH_DIM + 255) / 256), blk, 0, stream>>>(
      W_pe, wpe_t, IN_DIM, PATH_DIM);
  k_transpose_bf16<<<dim3((IN_DIM * HID_DIM + 255) / 256), blk, 0, stream>>>(
      W_hi, whi_t, IN_DIM, HID_DIM);
  k_transpose_bf16<<<dim3((FUSE_DIM * HID_DIM + 255) / 256), blk, 0, stream>>>(
      W1, w1_t, FUSE_DIM, HID_DIM);
  k_transpose_bf16<<<dim3((HID_DIM * HID_DIM + 255) / 256), blk, 0, stream>>>(
      W2, w2_t, HID_DIM, HID_DIM);
  k_transpose_bf16<<<dim3((HID_DIM * OUT_DIM + 255) / 256), blk, 0, stream>>>(
      W_ro, wro_t, HID_DIM, OUT_DIM);

  // ---- g = x0 @ W_pe + b_pe (fp32 out) ----
  k_gemm_bf16_wmma<<<dim3(BATCH / BM, PATH_DIM / BN), blk, 0, stream>>>(
      x0_bf, IN_DIM, wpe_t, IN_DIM, b_pe,
      BATCH, PATH_DIM, IN_DIM, /*mode=*/0, dt,
      zbuf, gbuf, inp_bf, FUSE_DIM);

  // ---- z0 = x0 @ W_hi + b_hi (fp32 z + bf16 mirror into inp[:, :1024]) ----
  k_gemm_bf16_wmma<<<dim3(BATCH / BM, HID_DIM / BN), blk, 0, stream>>>(
      x0_bf, IN_DIM, whi_t, IN_DIM, b_hi,
      BATCH, HID_DIM, IN_DIM, /*mode=*/1, dt,
      zbuf, gbuf, inp_bf, FUSE_DIM);

  // ---- 50 Euler steps ----
  for (int s = 1; s <= NSTEPS; ++s) {
    const float t = dt * (float)s;
    k_scale_g<<<dim3((BATCH * PATH_DIM) / 256), blk, 0, stream>>>(gbuf, inp_bf, t);

    // h = tanh(inp @ W1 + b1)  -> bf16
    k_gemm_bf16_wmma<<<dim3(BATCH / BM, HID_DIM / BN), blk, 0, stream>>>(
        inp_bf, FUSE_DIM, w1_t, FUSE_DIM, b1,
        BATCH, HID_DIM, FUSE_DIM, /*mode=*/2, dt,
        zbuf, gbuf, h_bf, HID_DIM);

    // z += dt*(h @ W2 + b2)    -> fp32 z + bf16 mirror into inp[:, :1024]
    k_gemm_bf16_wmma<<<dim3(BATCH / BM, HID_DIM / BN), blk, 0, stream>>>(
        h_bf, HID_DIM, w2_t, HID_DIM, b2,
        BATCH, HID_DIM, HID_DIM, /*mode=*/3, dt,
        zbuf, gbuf, inp_bf, FUSE_DIM);
  }

  // ---- out = z @ W_ro + b_ro (reads bf16 z mirror, fp32 out) ----
  k_gemm_bf16_wmma<<<dim3(BATCH / BM, OUT_DIM / BN), blk, 0, stream>>>(
      inp_bf, FUSE_DIM, wro_t, HID_DIM, b_ro,
      BATCH, OUT_DIM, HID_DIM, /*mode=*/0, dt,
      zbuf, out, h_bf, HID_DIM);
}
